// segmentation_model_58600533786805
// MI455X (gfx1250) — matrix operand order repaired and checked by reading, not compiled, not run on hardware
//
// PointNet++ segmentation for MI455X (gfx1250), wave32 + WMMA bf16.
//
// Roofline: ~8 GFLOP of edge-MLP GEMMs vs ~300MB of intermediate traffic that
// fits in the 192MB L2 -> bandwidth/latency bound. Therefore: bf16 activations
// (halve bytes), v_wmma_f32_16x16x32_bf16 for all linear layers, BatchNorm
// statistics fused into the GEMM epilogue (wave-level reduce + f32 atomics),
// neighbor search staged in LDS with one wave per query.
//
// v3: STATS/MASKED are template params (no per-element pointer checks), bias
// is folded into the WMMA C operand, and the m<E row guards are removed
// (every edge count in this model is a multiple of 16, so tiles are full).

#include <hip/hip_runtime.h>
#include <hip/hip_bf16.h>
#include <math.h>

typedef __attribute__((ext_vector_type(16))) __bf16 v16bf;
typedef __attribute__((ext_vector_type(8)))  __bf16 v8bf;
typedef __attribute__((ext_vector_type(8)))  float  v8f;

#define NBATCH 8
#define NP0    4096
#define NP1    1024
#define NP2    256
#define CH_IN  32
#define K_RAD  32
#define K_NN   64

// ---------------------------------------------------------------- utilities
__device__ __forceinline__ void atomicMaxFloat(float* addr, float val) {
    // standard sign-split trick; buffer is initialized to -inf
    if (val >= 0.0f) atomicMax((int*)addr, __float_as_int(val));
    else             atomicMin((unsigned int*)addr, (unsigned int)__float_as_int(val));
}

__global__ void fill_f32(float* __restrict__ p, long n, float v) {
    long i = (long)blockIdx.x * blockDim.x + threadIdx.x;
    long s = (long)gridDim.x * blockDim.x;
    for (; i < n; i += s) p[i] = v;
}

// ------------------------------------------------- weight pre-shuffle (bf16)
// Wp layout: [(kc*ntiles + nt)*32 + lane]*16 + j  with
//   half=lane>>4, col=lane&15, K = kc*32 + half*16 + j, N = nt*16 + col
// i.e. the per-lane v16bf B-fragment in the contiguous k order (k = 16*half + j) the A loads also use,
// so the GEMM loads B with one contiguous 32B vector load per lane.
__global__ void prep_w(const float* __restrict__ W, int Cin, int Cout,
                       int kchunks, int ntiles, __bf16* __restrict__ Wp) {
    int t = blockIdx.x * blockDim.x + threadIdx.x;
    int total = kchunks * ntiles * 512;
    if (t >= total) return;
    int j    = t & 15;
    int lane = (t >> 4) & 31;
    int rest = t >> 9;
    int nt   = rest % ntiles;
    int kc   = rest / ntiles;
    int half = lane >> 4, col = lane & 15;
    int kk = kc * 32 + half * 16 + j;
    int nn = nt * 16 + col;
    float v = (kk < Cin && nn < Cout) ? W[(size_t)kk * Cout + nn] : 0.0f;
    Wp[t] = (__bf16)v;
}

// -------------------------------------------------------------------- FPS
__global__ __launch_bounds__(256)
void fps_kernel(const float* __restrict__ pos, int N, int NS,
                int* __restrict__ outIdx, float* __restrict__ outPos) {
    __shared__ float d[4096];
    __shared__ float sval[256];
    __shared__ int   sidx[256];
    int b = blockIdx.x, t = threadIdx.x;
    const float* P = pos + (size_t)b * N * 3;
    float cx = P[0], cy = P[1], cz = P[2];
    for (int i = t; i < N; i += 256) {
        float dx = P[3*i] - cx, dy = P[3*i+1] - cy, dz = P[3*i+2] - cz;
        d[i] = dx*dx + dy*dy + dz*dz;
    }
    if (t == 0) {
        outIdx[b*NS] = 0;
        outPos[(size_t)(b*NS)*3+0] = cx;
        outPos[(size_t)(b*NS)*3+1] = cy;
        outPos[(size_t)(b*NS)*3+2] = cz;
    }
    __syncthreads();
    for (int s = 1; s < NS; ++s) {
        float best = -1.0f; int bi = N;
        for (int i = t; i < N; i += 256) {      // ascending i -> first-max kept
            float v = d[i];
            if (v > best) { best = v; bi = i; }
        }
        sval[t] = best; sidx[t] = bi;
        __syncthreads();
        for (int off = 128; off > 0; off >>= 1) {
            if (t < off) {
                float v2 = sval[t+off]; int i2 = sidx[t+off];
                if (v2 > sval[t] || (v2 == sval[t] && i2 < sidx[t])) {
                    sval[t] = v2; sidx[t] = i2;
                }
            }
            __syncthreads();
        }
        int nxt = sidx[0];
        float nx = P[3*nxt], ny = P[3*nxt+1], nz = P[3*nxt+2];
        if (t == 0) {
            outIdx[b*NS+s] = nxt;
            outPos[(size_t)(b*NS+s)*3+0] = nx;
            outPos[(size_t)(b*NS+s)*3+1] = ny;
            outPos[(size_t)(b*NS+s)*3+2] = nz;
        }
        __syncthreads();
        for (int i = t; i < N; i += 256) {
            float dx = P[3*i]-nx, dy = P[3*i+1]-ny, dz = P[3*i+2]-nz;
            d[i] = fminf(d[i], dx*dx + dy*dy + dz*dz);
        }
        __syncthreads();
    }
}

// ----------------------------------------------- neighbor selection (LDS)
// One wave32 per query; graph points staged in LDS (<=48KB).  Iterative
// selection with (d2,idx) threshold reproduces top_k-of-(-d2) within radius.
// r2 <= 0 -> pure kNN.
__global__ __launch_bounds__(256)
void nn_select(const float* __restrict__ qpos, const float* __restrict__ pos,
               int nq, int N, int K, float r2,
               int* __restrict__ nbr, float* __restrict__ cntOut) {
    __shared__ float px[4096], py[4096], pz[4096];
    int blocksPerGraph = nq / 8;
    int b  = blockIdx.x / blocksPerGraph;
    int q0 = (blockIdx.x % blocksPerGraph) * 8;
    const float* P = pos + (size_t)b * N * 3;
    for (int i = threadIdx.x; i < N; i += blockDim.x) {
        px[i] = P[3*i]; py[i] = P[3*i+1]; pz[i] = P[3*i+2];
    }
    __syncthreads();
    int wave = threadIdx.x >> 5, lane = threadIdx.x & 31;
    int gq = b * nq + q0 + wave;
    float qx = qpos[3*gq], qy = qpos[3*gq+1], qz = qpos[3*gq+2];
    float lastD = -1.0f; int lastI = -1;
    int found = 0;
    for (int k = 0; k < K; ++k) {
        float bd = 3.4e38f; int bi = 0x7fffffff;
        for (int i = lane; i < N; i += 32) {
            float dx = px[i]-qx, dy = py[i]-qy, dz = pz[i]-qz;
            float d2 = dx*dx + dy*dy + dz*dz;
            bool cand = (d2 > lastD) || (d2 == lastD && i > lastI);
            if (r2 > 0.0f && d2 > r2) cand = false;
            if (cand && (d2 < bd || (d2 == bd && i < bi))) { bd = d2; bi = i; }
        }
        for (int off = 16; off > 0; off >>= 1) {   // wave32 (d2,idx) min-reduce
            float od = __shfl_xor(bd, off);
            int   oi = __shfl_xor(bi, off);
            if (od < bd || (od == bd && oi < bi)) { bd = od; bi = oi; }
        }
        bool ok = (bd < 3.0e38f);
        nbr[(size_t)gq * K + k] = ok ? bi : -1;
        if (ok) { lastD = bd; lastI = bi; ++found; }
        else    { lastD = 3.4e38f; }
    }
    if (cntOut && lane == 0) atomicAdd(cntOut, (float)found);
}

// ----------------------------------------------- edge feature gathers (bf16)
__global__ __launch_bounds__(256)
void sa_gather(const float* __restrict__ x, int C,
               const float* __restrict__ pos, const float* __restrict__ qpos,
               const int* __restrict__ nbr, int NQ, int nqPerG, int N, int K,
               int Kpad, __bf16* __restrict__ A) {
    int e = blockIdx.x * blockDim.x + threadIdx.x;
    if (e >= NQ * K) return;
    int q = e / K, b = q / nqPerG;
    int j = nbr[e];
    __bf16* a = A + (size_t)e * Kpad;
    if (j >= 0) {
        const float* xs = x   + ((size_t)b * N + j) * C;
        const float* ps = pos + ((size_t)b * N + j) * 3;
        for (int c = 0; c < C; ++c) a[c] = (__bf16)xs[c];
        a[C]   = (__bf16)(ps[0] - qpos[3*q]);
        a[C+1] = (__bf16)(ps[1] - qpos[3*q+1]);
        a[C+2] = (__bf16)(ps[2] - qpos[3*q+2]);
    } else {
        for (int c = 0; c < C + 3; ++c) a[c] = (__bf16)0.0f;
    }
    for (int c = C + 3; c < Kpad; ++c) a[c] = (__bf16)0.0f;
}

__global__ __launch_bounds__(256)
void up_gather(const float* __restrict__ xc, int C,
               const float* __restrict__ posc, const float* __restrict__ posf,
               const int* __restrict__ nbr, int ncPerG, int nfPerG, int K,
               int Kpad, __bf16* __restrict__ A, int* __restrict__ dst,
               int E0, int Etot) {
    int e = blockIdx.x * blockDim.x + threadIdx.x;
    if (e >= Etot) return;
    int ci, d; float rx, ry, rz;
    if (e < E0) {
        ci = e / K;
        int b = ci / ncPerG;
        int gf = b * nfPerG + nbr[e];
        rx = posc[3*ci]   - posf[3*gf];
        ry = posc[3*ci+1] - posf[3*gf+1];
        rz = posc[3*ci+2] - posf[3*gf+2];
        d = gf;
    } else {                                  // global self loops, dst = i
        int i = e - E0;
        ci = i; d = i;
        rx = posc[3*i]   - posf[3*i];
        ry = posc[3*i+1] - posf[3*i+1];
        rz = posc[3*i+2] - posf[3*i+2];
    }
    __bf16* a = A + (size_t)e * Kpad;
    const float* xs = xc + (size_t)ci * C;
    for (int c = 0; c < C; ++c) a[c] = (__bf16)xs[c];
    a[C]   = (__bf16)rx;
    a[C+1] = (__bf16)ry;
    a[C+2] = (__bf16)rz;
    for (int c = C + 3; c < Kpad; ++c) a[c] = (__bf16)0.0f;
    dst[e] = d;
}

// ---------------------------------------------------------- WMMA edge GEMM
// One wave computes a 16-row x (NTL*16)-col slab of H = A @ W + bias.
// A: bf16 [E x lda] row-major, lda = 32*KCH; E must be a multiple of 16.
// Wp: pre-shuffled B-fragments.  bias is folded into the C operand (every
// accumulator row of a lane has the same output column, so C = bias
// broadcast gives A@W+b straight out of the MMA chain).
// STATS: fused BatchNorm sum/sumsq per channel (in-wave row reduce + 2
// atomics per lane); MASKED: only rows with mask[m]>=0 contribute.
template<int KCH, int NTL, bool STATS, bool MASKED>
__global__ __launch_bounds__(256)
void gemm_edges(const __bf16* __restrict__ A, int lda,
                const __bf16* __restrict__ Wp, const float* __restrict__ bias,
                float* __restrict__ H, int E, int Nout,
                const int* __restrict__ mask, float* __restrict__ stats) {
    int wave = threadIdx.x >> 5;
    int lane = threadIdx.x & 31;
    int m0 = (blockIdx.x * 8 + wave) * 16;
    if (m0 >= E) return;                       // wave-uniform: EXEC stays full
    int half = lane >> 4, row = lane & 15;
    int col0 = lane & 15;

    v8f acc[NTL];
#pragma unroll
    for (int nt = 0; nt < NTL; ++nt) {
        int c = nt * 16 + col0;
        float bv = (c < Nout) ? bias[c] : 0.0f;
#pragma unroll
        for (int r = 0; r < 8; ++r) acc[nt][r] = bv;
    }

    // A-fragment: lanes 0-15 hold K = kc*32+0..15,
    // lanes 16-31 hold K = kc*32+16..31: two 16B loads.
    const __bf16* ap = A + (size_t)(m0 + row) * lda + half * 16;
#pragma unroll
    for (int kc = 0; kc < KCH; ++kc) {
        v8bf lo = *(const v8bf*)(ap + kc * 32);
        v8bf hi = *(const v8bf*)(ap + kc * 32 + 8);
        v16bf a = __builtin_shufflevector(lo, hi, 0,1,2,3,4,5,6,7,
                                                  8,9,10,11,12,13,14,15);
#pragma unroll
        for (int nt = 0; nt < NTL; ++nt) {
            v16bf bfrag = *(const v16bf*)(Wp +
                (((size_t)kc * NTL + nt) * 32 + lane) * 16);
            acc[nt] = __builtin_amdgcn_wmma_f32_16x16x32_bf16(
                false, a, false, bfrag, (short)0, acc[nt], false, false);
        }
    }

    // Epilogue: D layout = row (lane/16)*8 + r, col = lane&15.
    int mbase = m0 + half * 8;
#pragma unroll
    for (int nt = 0; nt < NTL; ++nt) {
        int c = nt * 16 + col0;
        float s = 0.0f, s2 = 0.0f;
        if (c < Nout) {
#pragma unroll
            for (int r = 0; r < 8; ++r) {
                float v = acc[nt][r];
                H[(size_t)(mbase + r) * Nout + c] = v;
                if (STATS) {
                    bool valid = MASKED ? (mask[mbase + r] >= 0) : true;
                    if (valid) { s += v; s2 += v * v; }
                }
            }
        }
        if (STATS) {
            s  += __shfl_xor(s, 16);           // combine the two row-halves
            s2 += __shfl_xor(s2, 16);
            if (half == 0 && c < Nout) {
                atomicAdd(&stats[c], s);
                atomicAdd(&stats[64 + c], s2);
            }
        }
    }
}

// -------------------------------------- BatchNorm (training stats) + act
// act: 0=sigmoid, 1=relu.  Writes bf16 K-padded operand for next GEMM and/or
// a compact f32 copy.
__global__ __launch_bounds__(256)
void bn_act(const float* __restrict__ H, int E, int C,
            const float* __restrict__ stats, const float* __restrict__ cntPtr,
            float staticCnt, int act,
            __bf16* __restrict__ outA, int lda, float* __restrict__ outF) {
    int e = blockIdx.x * blockDim.x + threadIdx.x;
    if (e >= E) return;
    float cnt = cntPtr ? *cntPtr : staticCnt;
    cnt = fmaxf(cnt, 1.0f);
    for (int c = 0; c < C; ++c) {
        float mu  = stats[c] / cnt;
        float var = fmaxf(stats[64 + c] / cnt - mu * mu, 0.0f);
        float v = (H[(size_t)e * C + c] - mu) * rsqrtf(var + 1e-5f);
        v = (act == 0) ? (1.0f / (1.0f + __expf(-v))) : fmaxf(v, 0.0f);
        if (outA) outA[(size_t)e * lda + c] = (__bf16)v;
        if (outF) outF[(size_t)e * C + c] = v;
    }
    if (outA)
        for (int c = C; c < lda; ++c) outA[(size_t)e * lda + c] = (__bf16)0.0f;
}

// --------------------------------------------------------- aggregations
__global__ __launch_bounds__(256)
void sa_aggregate(const float* __restrict__ H, const int* __restrict__ nbr,
                  int NQ, int K, int C, float* __restrict__ out) {
    int t = blockIdx.x * blockDim.x + threadIdx.x;
    if (t >= NQ * C) return;
    int q = t / C, c = t % C;
    float m = -3.4e38f;
    for (int k = 0; k < K; ++k)
        if (nbr[(size_t)q * K + k] >= 0)
            m = fmaxf(m, H[((size_t)q * K + k) * C + c]);
    out[t] = fmaxf(m, 0.0f);   // relu; empty neighborhood -> 0 (matches ref)
}

__global__ __launch_bounds__(256)
void up_aggregate(const float* __restrict__ H, const int* __restrict__ dst,
                  int E, int C, float* __restrict__ agg) {
    int t = blockIdx.x * blockDim.x + threadIdx.x;
    if (t >= E * C) return;
    int e = t / C, c = t % C;
    atomicMaxFloat(&agg[(size_t)dst[e] * C + c], H[(size_t)e * C + c]);
}

// fix (-inf -> 0) + concat [agg, xf] -> bf16 padded operand for the lin layer
__global__ __launch_bounds__(256)
void concat_lin_gather(const float* __restrict__ agg, const float* __restrict__ xf,
                       int Nf, int Ca, int Cf, int Kpad, __bf16* __restrict__ A) {
    int i = blockIdx.x * blockDim.x + threadIdx.x;
    if (i >= Nf) return;
    __bf16* a = A + (size_t)i * Kpad;
    for (int c = 0; c < Ca; ++c) {
        float v = agg[(size_t)i * Ca + c];
        if (!(v > -1e37f)) v = 0.0f;
        a[c] = (__bf16)v;
    }
    for (int c = 0; c < Cf; ++c) a[Ca + c] = (__bf16)xf[(size_t)i * Cf + c];
    for (int c = Ca + Cf; c < Kpad; ++c) a[c] = (__bf16)0.0f;
}

// --------------------------------------------- final sigmoid + tuple output
__global__ __launch_bounds__(256)
void final_out(const float* __restrict__ Hfc, const float* __restrict__ pos,
               int BN, float* __restrict__ out) {
    int i = blockIdx.x * blockDim.x + threadIdx.x;
    if (i < BN)     out[i] = 1.0f / (1.0f + __expf(-Hfc[i]));
    if (i < BN * 3) out[BN + i] = pos[i];
    if (i < BN)     ((int*)out)[BN * 4 + i] = i / NP0;
}

// =========================================================== orchestration
extern "C" void kernel_launch(void* const* d_in, const int* in_sizes, int n_in,
                              void* d_out, int out_size, void* d_ws, size_t ws_size,
                              hipStream_t stream) {
    (void)in_sizes; (void)n_in; (void)out_size; (void)ws_size;
    const float* x   = (const float*)d_in[0];
    const float* pos = (const float*)d_in[1];

    // ---- workspace bump allocator (256B aligned)
    char* ws = (char*)d_ws;
    size_t off = 0;
    auto alloc = [&](size_t bytes) -> void* {
        off = (off + 255) & ~(size_t)255;
        void* p = ws + off;
        off += bytes;
        return p;
    };

    // ---- weights (d_in order follows setup_inputs pytree flattening)
    struct Lay { const float* W; const float* b; int Cin, Cout;
                 __bf16* wp; int kch, ntl; };
    Lay L[17] = {
        {(const float*)d_in[3],  (const float*)d_in[4],  35, 16, 0,0,0},
        {(const float*)d_in[5],  (const float*)d_in[6],  16, 16, 0,0,0},
        {(const float*)d_in[7],  (const float*)d_in[8],  16, 32, 0,0,0},
        {(const float*)d_in[9],  (const float*)d_in[10], 35, 32, 0,0,0},
        {(const float*)d_in[11], (const float*)d_in[12], 32, 32, 0,0,0},
        {(const float*)d_in[13], (const float*)d_in[14], 32, 64, 0,0,0},
        {(const float*)d_in[15], (const float*)d_in[16], 67, 64, 0,0,0},
        {(const float*)d_in[17], (const float*)d_in[18], 64, 64, 0,0,0},
        {(const float*)d_in[19], (const float*)d_in[20], 64, 32, 0,0,0},
        {(const float*)d_in[21], (const float*)d_in[22], 35, 32, 0,0,0},
        {(const float*)d_in[23], (const float*)d_in[24], 32, 32, 0,0,0},
        {(const float*)d_in[25], (const float*)d_in[26], 32, 32, 0,0,0},
        {(const float*)d_in[27], (const float*)d_in[28], 64, 32, 0,0,0},
        {(const float*)d_in[29], (const float*)d_in[30], 64, 32, 0,0,0},
        {(const float*)d_in[31], (const float*)d_in[32], 32, 32, 0,0,0},
        {(const float*)d_in[33], (const float*)d_in[34], 32, 32, 0,0,0},
        {(const float*)d_in[35], (const float*)d_in[36], 32,  1, 0,0,0},
    };

    const int E1  = NBATCH * NP1 * K_RAD;               // 262144
    const int E2  = NBATCH * NP2 * K_RAD;               // 65536
    const int EU1 = NBATCH * NP2 * K_NN + NBATCH * NP2; // 133120
    const int EU2 = NBATCH * NP1 * K_NN + NBATCH * NP1; // 532480
    const int NQ1 = NBATCH * NP1, NQ2 = NBATCH * NP2;
    const int NF1 = NBATCH * NP1, NF0 = NBATCH * NP0;

    int*   idx1  = (int*)alloc((size_t)NQ1 * 4);
    float* p1    = (float*)alloc((size_t)NQ1 * 3 * 4);
    int*   idx2  = (int*)alloc((size_t)NQ2 * 4);
    float* p2    = (float*)alloc((size_t)NQ2 * 3 * 4);
    int*   nbr1  = (int*)alloc((size_t)E1 * 4);
    int*   nbr2  = (int*)alloc((size_t)E2 * 4);
    int*   knn1  = (int*)alloc((size_t)NQ2 * K_NN * 4);
    int*   knn2  = (int*)alloc((size_t)NQ1 * K_NN * 4);
    int*   dst1  = (int*)alloc((size_t)EU1 * 4);
    int*   dst2  = (int*)alloc((size_t)EU2 * 4);
    float* stA   = (float*)alloc(12 * 128 * 4);   // 12 BN instances
    float* cnt1  = (float*)alloc(4);
    float* cnt2  = (float*)alloc(4);
    float* x1    = (float*)alloc((size_t)NQ1 * 32 * 4);
    float* x2    = (float*)alloc((size_t)NQ2 * 64 * 4);
    float* agg1  = (float*)alloc((size_t)NF1 * 32 * 4);
    float* agg2  = (float*)alloc((size_t)NF0 * 32 * 4);
    float* u1    = (float*)alloc((size_t)NF1 * 32 * 4);
    __bf16* ABUF = (__bf16*)alloc((size_t)EU2 * 64 * 2);  // 68MB
    float*  HBUF = (float*)alloc((size_t)EU2 * 32 * 4);   // 68MB

    auto st = [&](int i) { return stA + i * 128; };
    auto fill = [&](float* p, long n, float v) {
        int blocks = (int)((n + 255) / 256); if (blocks > 4096) blocks = 4096;
        fill_f32<<<blocks, 256, 0, stream>>>(p, n, v);
    };
    // static dispatch over the (kchunks, ntiles, stats, masked) shapes used
    auto gemm = [&](const __bf16* A, int lda, const Lay& l, float* H, int E,
                    const int* mask, float* stats) {
        int blocks = (E + 127) / 128;
        bool S = stats != nullptr, M = mask != nullptr;
#define GO(KC,NT,SS,MM) gemm_edges<KC,NT,SS,MM><<<blocks,256,0,stream>>>( \
            A, lda, l.wp, l.b, H, E, l.Cout, mask, stats)
        if (l.kch == 1 && l.ntl == 1) {
            if (S)           GO(1,1,true,true);   // SA1 layer1
            else             GO(1,1,false,false); // fc last (Nout=1)
        } else if (l.kch == 1 && l.ntl == 2) {
            if (S && M)      GO(1,2,true,true);   // SA2 layer1
            else if (S)      GO(1,2,true,false);  // up2 l1, fc0, fc1
            else             GO(1,2,false,false); // SA1 l2, up2 l2
        } else if (l.kch == 1 && l.ntl == 4) {
            GO(1,4,false,false);                  // SA2 layer2
        } else if (l.kch == 2 && l.ntl == 1) {
            GO(2,1,true,true);                    // SA1 layer0
        } else if (l.kch == 2 && l.ntl == 2) {
            if (S && M)      GO(2,2,true,true);   // SA2 layer0
            else if (S)      GO(2,2,true,false);  // up2 l0, lin1, lin2
            else             GO(2,2,false,false); // up1 l2
        } else if (l.kch == 2 && l.ntl == 4) {
            GO(2,4,true,false);                   // up1 layer1
        } else {
            GO(3,4,true,false);                   // up1 layer0
        }
#undef GO
    };
    auto bn = [&](float* H, int E, int C, float* stats, const float* cp,
                  float sc, int act, __bf16* oA, int lda, float* oF) {
        bn_act<<<(E + 255) / 256, 256, 0, stream>>>(
            H, E, C, stats, cp, sc, act, oA, lda, oF);
    };

    // ---- per-launch init (deterministic across graph replays)
    fill(stA, 12 * 128, 0.0f);
    fill(cnt1, 1, 0.0f);
    fill(cnt2, 1, 0.0f);
    fill(agg1, (long)NF1 * 32, -INFINITY);
    fill(agg2, (long)NF0 * 32, -INFINITY);
    for (int i = 0; i < 17; ++i) {
        L[i].kch = (L[i].Cin + 31) / 32;
        L[i].ntl = (L[i].Cout + 15) / 16;
        size_t n = (size_t)L[i].kch * L[i].ntl * 512;
        L[i].wp = (__bf16*)alloc(n * 2);
        prep_w<<<(int)((n + 255) / 256), 256, 0, stream>>>(
            L[i].W, L[i].Cin, L[i].Cout, L[i].kch, L[i].ntl, L[i].wp);
    }

    // ================= SA1: FPS -> radius(32,r=1) -> MLP[35,16,16,32] -> max
    fps_kernel<<<NBATCH, 256, 0, stream>>>(pos, NP0, NP1, idx1, p1);
    nn_select<<<NBATCH * (NP1 / 8), 256, 0, stream>>>(p1, pos, NP1, NP0,
                                                      K_RAD, 1.0f, nbr1, cnt1);
    sa_gather<<<(E1 + 255) / 256, 256, 0, stream>>>(x, CH_IN, pos, p1, nbr1,
                                                    NQ1, NP1, NP0, K_RAD, 64, ABUF);
    gemm(ABUF, 64, L[0], HBUF, E1, nbr1, st(0));
    bn(HBUF, E1, 16, st(0), cnt1, 0.0f, 0, ABUF, 32, nullptr);
    gemm(ABUF, 32, L[1], HBUF, E1, nbr1, st(1));
    bn(HBUF, E1, 16, st(1), cnt1, 0.0f, 0, ABUF, 32, nullptr);
    gemm(ABUF, 32, L[2], HBUF, E1, nullptr, nullptr);
    sa_aggregate<<<(NQ1 * 32 + 255) / 256, 256, 0, stream>>>(HBUF, nbr1, NQ1,
                                                             K_RAD, 32, x1);

    // ================= SA2: FPS -> radius(32,r=2) -> MLP[35,32,32,64] -> max
    fps_kernel<<<NBATCH, 256, 0, stream>>>(p1, NP1, NP2, idx2, p2);
    nn_select<<<NBATCH * (NP2 / 8), 256, 0, stream>>>(p2, p1, NP2, NP1,
                                                      K_RAD, 4.0f, nbr2, cnt2);
    sa_gather<<<(E2 + 255) / 256, 256, 0, stream>>>(x1, 32, p1, p2, nbr2,
                                                    NQ2, NP2, NP1, K_RAD, 64, ABUF);
    gemm(ABUF, 64, L[3], HBUF, E2, nbr2, st(2));
    bn(HBUF, E2, 32, st(2), cnt2, 0.0f, 0, ABUF, 32, nullptr);
    gemm(ABUF, 32, L[4], HBUF, E2, nbr2, st(3));
    bn(HBUF, E2, 32, st(3), cnt2, 0.0f, 0, ABUF, 32, nullptr);
    gemm(ABUF, 32, L[5], HBUF, E2, nullptr, nullptr);
    sa_aggregate<<<(NQ2 * 64 + 255) / 256, 256, 0, stream>>>(HBUF, nbr2, NQ2,
                                                             K_RAD, 64, x2);

    // ================= UP1: knn(64) -> MLP[67,64,64,32] -> seg-max -> lin+BN
    nn_select<<<NBATCH * (NP2 / 8), 256, 0, stream>>>(p2, p1, NP2, NP1,
                                                      K_NN, -1.0f, knn1, nullptr);
    up_gather<<<(EU1 + 255) / 256, 256, 0, stream>>>(x2, 64, p2, p1, knn1,
                                                     NP2, NP1, K_NN, 96, ABUF,
                                                     dst1, NQ2 * K_NN, EU1);
    gemm(ABUF, 96, L[6], HBUF, EU1, nullptr, st(4));
    bn(HBUF, EU1, 64, st(4), nullptr, (float)EU1, 0, ABUF, 64, nullptr);
    gemm(ABUF, 64, L[7], HBUF, EU1, nullptr, st(5));
    bn(HBUF, EU1, 64, st(5), nullptr, (float)EU1, 0, ABUF, 64, nullptr);
    gemm(ABUF, 64, L[8], HBUF, EU1, nullptr, nullptr);
    up_aggregate<<<(EU1 * 32 + 255) / 256, 256, 0, stream>>>(HBUF, dst1, EU1,
                                                             32, agg1);
    concat_lin_gather<<<(NF1 + 255) / 256, 256, 0, stream>>>(agg1, x1, NF1,
                                                             32, 32, 64, ABUF);
    gemm(ABUF, 64, L[12], HBUF, NF1, nullptr, st(6));
    bn(HBUF, NF1, 32, st(6), nullptr, (float)NF1, 1, nullptr, 0, u1);

    // ================= UP2: knn(64) -> MLP[35,32,32,32] -> seg-max -> lin+BN
    nn_select<<<NBATCH * (NP1 / 8), 256, 0, stream>>>(p1, pos, NP1, NP0,
                                                      K_NN, -1.0f, knn2, nullptr);
    up_gather<<<(EU2 + 255) / 256, 256, 0, stream>>>(u1, 32, p1, pos, knn2,
                                                     NP1, NP0, K_NN, 64, ABUF,
                                                     dst2, NQ1 * K_NN, EU2);
    gemm(ABUF, 64, L[9], HBUF, EU2, nullptr, st(7));
    bn(HBUF, EU2, 32, st(7), nullptr, (float)EU2, 0, ABUF, 32, nullptr);
    gemm(ABUF, 32, L[10], HBUF, EU2, nullptr, st(8));
    bn(HBUF, EU2, 32, st(8), nullptr, (float)EU2, 0, ABUF, 32, nullptr);
    gemm(ABUF, 32, L[11], HBUF, EU2, nullptr, nullptr);
    up_aggregate<<<(EU2 * 32 + 255) / 256, 256, 0, stream>>>(HBUF, dst2, EU2,
                                                             32, agg2);
    concat_lin_gather<<<(NF0 + 255) / 256, 256, 0, stream>>>(agg2, x, NF0,
                                                             32, 32, 64, ABUF);
    gemm(ABUF, 64, L[13], HBUF, NF0, nullptr, st(9));
    bn(HBUF, NF0, 32, st(9), nullptr, (float)NF0, 1, ABUF, 32, nullptr);

    // ================= FC head [32,32,32,1] + sigmoid, tuple output
    gemm(ABUF, 32, L[14], HBUF, NF0, nullptr, st(10));
    bn(HBUF, NF0, 32, st(10), nullptr, (float)NF0, 1, ABUF, 32, nullptr);
    gemm(ABUF, 32, L[15], HBUF, NF0, nullptr, st(11));
    bn(HBUF, NF0, 32, st(11), nullptr, (float)NF0, 1, ABUF, 32, nullptr);
    gemm(ABUF, 32, L[16], HBUF, NF0, nullptr, nullptr);
    final_out<<<(NF0 * 3 + 255) / 256, 256, 0, stream>>>(HBUF, pos, NF0,
                                                         (float*)d_out);
}
